// AttnNetwork_13022340841497
// MI455X (gfx1250) — compile-verified, matrix operand order checked by reading, not run on hardware
//
#include <hip/hip_runtime.h>
#include <hip/hip_bf16.h>
#include <math.h>

// ---------------- problem constants ----------------
constexpr int VDE_ = 30000, VEN_ = 30000;
constexpr int D_ = 620, H_ = 1000, M_ = 1000;
constexpr int B_ = 32, S_ = 20, T_ = 20;
constexpr int G4H = 4 * H_;          // 4000
constexpr int DPAD = 640;            // D padded to mult of 32
constexpr int HPAD = 1024;           // H padded to mult of 32
constexpr int N4H_PAD = 4096;        // 4H padded to mult of 256 (block tile)
constexpr int NH_PAD  = 1024;        // H  padded to mult of 256
constexpr int N2M_PAD = 2048;        // 2M padded to mult of 256
constexpr int NV_PAD  = 30208;       // VEN padded to mult of 256

// ---------------- WMMA types ----------------
typedef __attribute__((ext_vector_type(16))) __bf16 v16bf;
typedef __attribute__((ext_vector_type(8)))  __bf16 v8bf;
typedef __attribute__((ext_vector_type(8)))  float  v8f;

union FragU { v16bf v; v8bf h[2]; };

__device__ __forceinline__ float sigf(float x) { return 1.0f / (1.0f + __expf(-x)); }

// =====================================================================
// Generic GEMM:  C[M,N] = A[M,K](bf16) * W[N,K](bf16)^T  (+bias) (+Cin)
// A row-major lda (K-contiguous); W row-major ldw (K-contiguous).
// K multiple of 32. Grid: x = Npad/256, y = ceil(M/32). Block 256 = 8 waves.
// Wave (wrow,wcol) computes rows [row0,row0+16) x cols [col0,col0+64).
// =====================================================================
__global__ void gemm_bf16_kernel(const __hip_bfloat16* __restrict__ A, int lda,
                                 const __hip_bfloat16* __restrict__ W, int ldw,
                                 float* __restrict__ C, int ldc,
                                 const float* __restrict__ Cin, int ldcin,
                                 const float* __restrict__ bias,
                                 int M, int N, int K)
{
    const int lane = threadIdx.x & 31;
    const int wave = threadIdx.x >> 5;
    const int wrow = wave >> 2;                 // 0..1
    const int wcol = wave & 3;                  // 0..3
    const int row0 = blockIdx.y * 32 + wrow * 16;
    const int col0 = blockIdx.x * 256 + wcol * 64;
    if (row0 >= M) return;

    const int l15  = lane & 15;
    const int half = lane >> 4;                 // 0: K 0..7/16..23, 1: K 8..15/24..31
    const int ksub = half << 3;                 // 0 or 8

    const __hip_bfloat16* aptr = A + (size_t)(row0 + l15) * lda + ksub;

    v8f acc0 = {}, acc1 = {}, acc2 = {}, acc3 = {};

    const __hip_bfloat16* b0 = W + (size_t)(col0 + 0 * 16 + l15) * ldw + ksub;
    const __hip_bfloat16* b1 = W + (size_t)(col0 + 1 * 16 + l15) * ldw + ksub;
    const __hip_bfloat16* b2 = W + (size_t)(col0 + 2 * 16 + l15) * ldw + ksub;
    const __hip_bfloat16* b3 = W + (size_t)(col0 + 3 * 16 + l15) * ldw + ksub;

    for (int k0 = 0; k0 < K; k0 += 32) {
        FragU a, b;
        a.h[0] = *(const v8bf*)(aptr + k0);
        a.h[1] = *(const v8bf*)(aptr + k0 + 16);

        b.h[0] = *(const v8bf*)(b0 + k0);
        b.h[1] = *(const v8bf*)(b0 + k0 + 16);
        acc0 = __builtin_amdgcn_wmma_f32_16x16x32_bf16(false, a.v, false, b.v,
                                                       (short)0, acc0, false, false);
        b.h[0] = *(const v8bf*)(b1 + k0);
        b.h[1] = *(const v8bf*)(b1 + k0 + 16);
        acc1 = __builtin_amdgcn_wmma_f32_16x16x32_bf16(false, a.v, false, b.v,
                                                       (short)0, acc1, false, false);
        b.h[0] = *(const v8bf*)(b2 + k0);
        b.h[1] = *(const v8bf*)(b2 + k0 + 16);
        acc2 = __builtin_amdgcn_wmma_f32_16x16x32_bf16(false, a.v, false, b.v,
                                                       (short)0, acc2, false, false);
        b.h[0] = *(const v8bf*)(b3 + k0);
        b.h[1] = *(const v8bf*)(b3 + k0 + 16);
        acc3 = __builtin_amdgcn_wmma_f32_16x16x32_bf16(false, a.v, false, b.v,
                                                       (short)0, acc3, false, false);
    }

    // C/D layout: VGPR r, lane: row = row0 + r + 8*half, col = col0 + t*16 + l15
    v8f accs[4] = {acc0, acc1, acc2, acc3};
#pragma unroll
    for (int t = 0; t < 4; ++t) {
        const int col = col0 + t * 16 + l15;
        if (col >= N) continue;
        const float badd = bias ? bias[col] : 0.0f;
#pragma unroll
        for (int r = 0; r < 8; ++r) {
            const int row = row0 + r + 8 * half;
            if (row < M) {
                float v = accs[t][r] + badd;
                if (Cin) v += Cin[(size_t)row * ldcin + col];
                C[(size_t)row * ldc + col] = v;
            }
        }
    }
}

// ============ weight fp32 -> zero-padded bf16 [Npad, Kpad] ============
__global__ void cvt_weight_kernel(const float* __restrict__ src,
                                  __hip_bfloat16* __restrict__ dst,
                                  int N, int K, int Npad, int Kpad)
{
    int idx = blockIdx.x * blockDim.x + threadIdx.x;
    if (idx >= Npad * Kpad) return;
    int k = idx % Kpad, n = idx / Kpad;
    float v = (n < N && k < K) ? src[(size_t)n * K + k] : 0.0f;
    dst[idx] = __float2bfloat16(v);
}

// ============ embedding gather -> zero-padded bf16 rows ============
// row r = b*rowsPerB + j maps to ids[b*srcStride + j]
__global__ void gather_emb_kernel(const float* __restrict__ emb,
                                  const int* __restrict__ ids,
                                  __hip_bfloat16* __restrict__ dst,
                                  int rows, int rowsPerB, int srcStride)
{
    int idx = blockIdx.x * blockDim.x + threadIdx.x;
    if (idx >= rows * DPAD) return;
    int k = idx % DPAD, r = idx / DPAD;
    int b = r / rowsPerB, j = r % rowsPerB;
    int id = ids[b * srcStride + j];
    float v = (k < D_) ? emb[(size_t)id * D_ + k] : 0.0f;
    dst[idx] = __float2bfloat16(v);
}

// ============ zero LSTM state ============
__global__ void zero_state_kernel(float* __restrict__ h, float* __restrict__ c,
                                  __hip_bfloat16* __restrict__ h_bf)
{
    int i = blockIdx.x * blockDim.x + threadIdx.x;
    if (i < B_ * H_) { h[i] = 0.0f; c[i] = 0.0f; }
    if (i < B_ * HPAD) h_bf[i] = __float2bfloat16(0.0f);
}

// ============ LSTM pointwise cell (gate order i,f,g,o) ============
__global__ void lstm_cell_kernel(const float* __restrict__ gates,
                                 float* __restrict__ h, float* __restrict__ c,
                                 __hip_bfloat16* __restrict__ h_bf,
                                 float* __restrict__ hist,
                                 __hip_bfloat16* __restrict__ hist_bf,
                                 int t, int steps)
{
    int idx = blockIdx.x * blockDim.x + threadIdx.x;
    if (idx >= B_ * HPAD) return;
    int b = idx / HPAD, j = idx % HPAD;
    size_t hrow = (size_t)(b * steps + t);
    if (j < H_) {
        size_t g0 = (size_t)b * G4H;
        float gi = gates[g0 + j];
        float gf = gates[g0 + H_ + j];
        float gg = gates[g0 + 2 * H_ + j];
        float go = gates[g0 + 3 * H_ + j];
        float cn = sigf(gf) * c[b * H_ + j] + sigf(gi) * tanhf(gg);
        float hn = sigf(go) * tanhf(cn);
        c[b * H_ + j] = cn;
        h[b * H_ + j] = hn;
        h_bf[b * HPAD + j] = __float2bfloat16(hn);
        hist[hrow * H_ + j] = hn;
        hist_bf[hrow * HPAD + j] = __float2bfloat16(hn);
    } else {
        h_bf[b * HPAD + j] = __float2bfloat16(0.0f);
        hist_bf[hrow * HPAD + j] = __float2bfloat16(0.0f);
    }
}

// ============ attention scores: one wave per (b,t,s) ============
__global__ void attn_scores_kernel(const float* __restrict__ Ua,
                                   const float* __restrict__ Wa,
                                   const float* __restrict__ Va_w,
                                   const float* __restrict__ Va_b,
                                   float* __restrict__ scores)
{
    int gw = (blockIdx.x * blockDim.x + threadIdx.x) >> 5;
    int lane = threadIdx.x & 31;
    if (gw >= B_ * T_ * S_) return;
    int s = gw % S_;
    int t = (gw / S_) % T_;
    int b = gw / (S_ * T_);
    const float* ua = Ua + (size_t)(b * S_ + s) * H_;
    const float* wa = Wa + (size_t)(b * T_ + t) * H_;
    float sum = 0.0f;
    for (int hh = lane; hh < H_; hh += 32)
        sum += Va_w[hh] * tanhf(ua[hh] + wa[hh]);
#pragma unroll
    for (int off = 16; off > 0; off >>= 1)
        sum += __shfl_down(sum, off, 32);
    if (lane == 0)
        scores[(size_t)(b * T_ + t) * S_ + s] = sum + Va_b[0];
}

// ============ softmax over s (row length S) ============
__global__ void softmax_s_kernel(const float* __restrict__ scores,
                                 float* __restrict__ attn)
{
    int r = blockIdx.x * blockDim.x + threadIdx.x;
    if (r >= B_ * T_) return;
    const float* in = scores + (size_t)r * S_;
    float* out = attn + (size_t)r * S_;
    float mx = in[0];
    for (int s = 1; s < S_; ++s) mx = fmaxf(mx, in[s]);
    float sum = 0.0f;
    for (int s = 0; s < S_; ++s) { float e = __expf(in[s] - mx); out[s] = e; sum += e; }
    float inv = 1.0f / sum;
    for (int s = 0; s < S_; ++s) out[s] *= inv;
}

// ============ context[b,t,:] = sum_s attn * enc_h, to padded bf16 ============
__global__ void context_kernel(const float* __restrict__ attn,
                               const float* __restrict__ enc_h,
                               __hip_bfloat16* __restrict__ ctx_bf)
{
    int idx = blockIdx.x * blockDim.x + threadIdx.x;
    if (idx >= B_ * T_ * HPAD) return;
    int j = idx % HPAD, bt = idx / HPAD;
    if (j < H_) {
        int b = bt / T_;
        const float* a = attn + (size_t)bt * S_;
        float sum = 0.0f;
        for (int s = 0; s < S_; ++s)
            sum += a[s] * enc_h[(size_t)(b * S_ + s) * H_ + j];
        ctx_bf[(size_t)bt * HPAD + j] = __float2bfloat16(sum);
    } else {
        ctx_bf[(size_t)bt * HPAD + j] = __float2bfloat16(0.0f);
    }
}

// ============ maxout: t_max[.,m] = max(u[.,2m], u[.,2m+1]) -> padded bf16 ===
__global__ void maxout_kernel(const float* __restrict__ u,
                              __hip_bfloat16* __restrict__ tmax_bf)
{
    int idx = blockIdx.x * blockDim.x + threadIdx.x;
    if (idx >= B_ * T_ * HPAD) return;
    int m = idx % HPAD, bt = idx / HPAD;
    float v = 0.0f;
    if (m < M_) {
        const float* ur = u + (size_t)bt * (2 * M_);
        v = fmaxf(ur[2 * m], ur[2 * m + 1]);
    }
    tmax_bf[(size_t)bt * HPAD + m] = __float2bfloat16(v);
}

// ============ per-row log-softmax + NLL (block per (b,t) row) ============
__global__ void loss_row_kernel(const float* __restrict__ logits,
                                const int* __restrict__ y,
                                const float* __restrict__ Wb_unused,
                                float* __restrict__ nll)
{
    __shared__ float red[256];
    const int row = blockIdx.x;                    // 0..B*T-1
    const float* lr = logits + (size_t)row * VEN_;
    float mx = -INFINITY;
    for (int v = threadIdx.x; v < VEN_; v += 256) mx = fmaxf(mx, lr[v]);
    red[threadIdx.x] = mx;
    __syncthreads();
    for (int s = 128; s > 0; s >>= 1) {
        if (threadIdx.x < s) red[threadIdx.x] = fmaxf(red[threadIdx.x], red[threadIdx.x + s]);
        __syncthreads();
    }
    mx = red[0];
    __syncthreads();
    float sum = 0.0f;
    for (int v = threadIdx.x; v < VEN_; v += 256) sum += __expf(lr[v] - mx);
    red[threadIdx.x] = sum;
    __syncthreads();
    for (int s = 128; s > 0; s >>= 1) {
        if (threadIdx.x < s) red[threadIdx.x] += red[threadIdx.x + s];
        __syncthreads();
    }
    if (threadIdx.x == 0) {
        int b = row / T_, t = row % T_;
        int lbl = y[b * (T_ + 1) + t + 1];
        nll[row] = -(lr[lbl] - mx - logf(red[0]));
    }
}

// ============ fixed-order final reduction (deterministic) ============
__global__ void loss_final_kernel(const float* __restrict__ nll, float* __restrict__ out)
{
    if (blockIdx.x == 0 && threadIdx.x == 0) {
        float s = 0.0f;
        for (int i = 0; i < B_ * T_; ++i) s += nll[i];
        out[0] = s / (float)B_;
    }
}

// =====================================================================
extern "C" void kernel_launch(void* const* d_in, const int* in_sizes, int n_in,
                              void* d_out, int out_size, void* d_ws, size_t ws_size,
                              hipStream_t stream)
{
    const int*   x       = (const int*)  d_in[0];
    const int*   y       = (const int*)  d_in[1];
    const float* emb_de  = (const float*)d_in[2];
    const float* emb_en  = (const float*)d_in[3];
    const float* enc_Wih = (const float*)d_in[4];
    const float* enc_Whh = (const float*)d_in[5];
    const float* enc_b   = (const float*)d_in[6];
    const float* dec_Wih = (const float*)d_in[7];
    const float* dec_Whh = (const float*)d_in[8];
    const float* dec_b   = (const float*)d_in[9];
    const float* Wa_w    = (const float*)d_in[10];
    const float* Wa_b    = (const float*)d_in[11];
    const float* Ua_w    = (const float*)d_in[12];
    const float* Ua_b    = (const float*)d_in[13];
    const float* Va_w    = (const float*)d_in[14];
    const float* Va_b    = (const float*)d_in[15];
    const float* U_w     = (const float*)d_in[16];
    const float* U_b     = (const float*)d_in[17];
    const float* V_w     = (const float*)d_in[18];
    const float* V_b     = (const float*)d_in[19];
    const float* C_w     = (const float*)d_in[20];
    const float* C_b     = (const float*)d_in[21];
    const float* W_w     = (const float*)d_in[22];
    const float* W_b     = (const float*)d_in[23];
    float* out = (float*)d_out;

    // ---- workspace allocator ----
    char* base = (char*)d_ws;
    size_t off = 0;
    auto alloc = [&](size_t bytes) -> void* {
        off = (off + 255) & ~(size_t)255;
        void* p = base + off;
        off += bytes;
        return p;
    };
    typedef __hip_bfloat16 bf;

    bf* enc_Wih_bf = (bf*)alloc((size_t)N4H_PAD * DPAD * 2);
    bf* enc_Whh_bf = (bf*)alloc((size_t)N4H_PAD * HPAD * 2);
    bf* dec_Wih_bf = (bf*)alloc((size_t)N4H_PAD * DPAD * 2);
    bf* dec_Whh_bf = (bf*)alloc((size_t)N4H_PAD * HPAD * 2);
    bf* Ua_w_bf    = (bf*)alloc((size_t)NH_PAD  * HPAD * 2);
    bf* Wa_w_bf    = (bf*)alloc((size_t)NH_PAD  * HPAD * 2);
    bf* U_w_bf     = (bf*)alloc((size_t)N2M_PAD * HPAD * 2);
    bf* V_w_bf     = (bf*)alloc((size_t)N2M_PAD * DPAD * 2);
    bf* C_w_bf     = (bf*)alloc((size_t)N2M_PAD * HPAD * 2);
    bf* W_w_bf     = (bf*)alloc((size_t)NV_PAD  * HPAD * 2);

    bf*    e_de_bf  = (bf*)alloc((size_t)B_ * S_ * DPAD * 2);
    bf*    e_en_bf  = (bf*)alloc((size_t)B_ * T_ * DPAD * 2);
    float* xp_enc   = (float*)alloc((size_t)B_ * S_ * G4H * 4);
    float* xp_dec   = (float*)alloc((size_t)B_ * T_ * G4H * 4);
    float* h_st     = (float*)alloc((size_t)B_ * H_ * 4);
    float* c_st     = (float*)alloc((size_t)B_ * H_ * 4);
    bf*    h_bf     = (bf*)alloc((size_t)B_ * HPAD * 2);
    float* gates    = (float*)alloc((size_t)B_ * G4H * 4);
    float* enc_h    = (float*)alloc((size_t)B_ * S_ * H_ * 4);
    bf*    enc_h_bf = (bf*)alloc((size_t)B_ * S_ * HPAD * 2);
    float* dec_h    = (float*)alloc((size_t)B_ * T_ * H_ * 4);
    bf*    dec_h_bf = (bf*)alloc((size_t)B_ * T_ * HPAD * 2);
    float* Ua       = (float*)alloc((size_t)B_ * S_ * H_ * 4);
    float* Wa       = (float*)alloc((size_t)B_ * T_ * H_ * 4);
    float* scores   = (float*)alloc((size_t)B_ * T_ * S_ * 4);
    float* attn     = (float*)alloc((size_t)B_ * T_ * S_ * 4);
    bf*    ctx_bf   = (bf*)alloc((size_t)B_ * T_ * HPAD * 2);
    float* u_buf    = (float*)alloc((size_t)B_ * T_ * 2 * M_ * 4);
    bf*    tmax_bf  = (bf*)alloc((size_t)B_ * T_ * HPAD * 2);
    float* logits   = (float*)alloc((size_t)B_ * T_ * VEN_ * 4);
    float* nll      = (float*)alloc((size_t)B_ * T_ * 4);
    (void)ws_size; (void)in_sizes; (void)n_in; (void)out_size;

    const dim3 blk(256);
    auto cgrid = [](size_t n) { return dim3((unsigned)((n + 255) / 256)); };

    // ---- weight conversions to padded bf16 ----
    hipLaunchKernelGGL(cvt_weight_kernel, cgrid((size_t)N4H_PAD * DPAD), blk, 0, stream,
                       enc_Wih, enc_Wih_bf, G4H, D_, N4H_PAD, DPAD);
    hipLaunchKernelGGL(cvt_weight_kernel, cgrid((size_t)N4H_PAD * HPAD), blk, 0, stream,
                       enc_Whh, enc_Whh_bf, G4H, H_, N4H_PAD, HPAD);
    hipLaunchKernelGGL(cvt_weight_kernel, cgrid((size_t)N4H_PAD * DPAD), blk, 0, stream,
                       dec_Wih, dec_Wih_bf, G4H, D_, N4H_PAD, DPAD);
    hipLaunchKernelGGL(cvt_weight_kernel, cgrid((size_t)N4H_PAD * HPAD), blk, 0, stream,
                       dec_Whh, dec_Whh_bf, G4H, H_, N4H_PAD, HPAD);
    hipLaunchKernelGGL(cvt_weight_kernel, cgrid((size_t)NH_PAD * HPAD), blk, 0, stream,
                       Ua_w, Ua_w_bf, H_, H_, NH_PAD, HPAD);
    hipLaunchKernelGGL(cvt_weight_kernel, cgrid((size_t)NH_PAD * HPAD), blk, 0, stream,
                       Wa_w, Wa_w_bf, H_, H_, NH_PAD, HPAD);
    hipLaunchKernelGGL(cvt_weight_kernel, cgrid((size_t)N2M_PAD * HPAD), blk, 0, stream,
                       U_w, U_w_bf, 2 * M_, H_, N2M_PAD, HPAD);
    hipLaunchKernelGGL(cvt_weight_kernel, cgrid((size_t)N2M_PAD * DPAD), blk, 0, stream,
                       V_w, V_w_bf, 2 * M_, D_, N2M_PAD, DPAD);
    hipLaunchKernelGGL(cvt_weight_kernel, cgrid((size_t)N2M_PAD * HPAD), blk, 0, stream,
                       C_w, C_w_bf, 2 * M_, H_, N2M_PAD, HPAD);
    hipLaunchKernelGGL(cvt_weight_kernel, cgrid((size_t)NV_PAD * HPAD), blk, 0, stream,
                       W_w, W_w_bf, VEN_, H_, NV_PAD, HPAD);

    // ---- embedding gathers ----
    hipLaunchKernelGGL(gather_emb_kernel, cgrid((size_t)B_ * S_ * DPAD), blk, 0, stream,
                       emb_de, x, e_de_bf, B_ * S_, S_, S_);
    hipLaunchKernelGGL(gather_emb_kernel, cgrid((size_t)B_ * T_ * DPAD), blk, 0, stream,
                       emb_en, y, e_en_bf, B_ * T_, T_, T_ + 1);

    // ---- input projections xp = e @ Wih^T + b ----
    hipLaunchKernelGGL(gemm_bf16_kernel, dim3(N4H_PAD / 256, (B_ * S_) / 32), blk, 0, stream,
                       e_de_bf, DPAD, enc_Wih_bf, DPAD, xp_enc, G4H,
                       (const float*)nullptr, 0, enc_b, B_ * S_, G4H, DPAD);
    hipLaunchKernelGGL(gemm_bf16_kernel, dim3(N4H_PAD / 256, (B_ * T_) / 32), blk, 0, stream,
                       e_en_bf, DPAD, dec_Wih_bf, DPAD, xp_dec, G4H,
                       (const float*)nullptr, 0, dec_b, B_ * T_, G4H, DPAD);

    // ---- encoder LSTM ----
    hipLaunchKernelGGL(zero_state_kernel, cgrid((size_t)B_ * HPAD), blk, 0, stream,
                       h_st, c_st, h_bf);
    for (int t = 0; t < S_; ++t) {
        hipLaunchKernelGGL(gemm_bf16_kernel, dim3(N4H_PAD / 256, 1), blk, 0, stream,
                           h_bf, HPAD, enc_Whh_bf, HPAD, gates, G4H,
                           xp_enc + (size_t)t * G4H, S_ * G4H,
                           (const float*)nullptr, B_, G4H, HPAD);
        hipLaunchKernelGGL(lstm_cell_kernel, cgrid((size_t)B_ * HPAD), blk, 0, stream,
                           gates, h_st, c_st, h_bf, enc_h, enc_h_bf, t, S_);
    }

    // ---- decoder LSTM ----
    hipLaunchKernelGGL(zero_state_kernel, cgrid((size_t)B_ * HPAD), blk, 0, stream,
                       h_st, c_st, h_bf);
    for (int t = 0; t < T_; ++t) {
        hipLaunchKernelGGL(gemm_bf16_kernel, dim3(N4H_PAD / 256, 1), blk, 0, stream,
                           h_bf, HPAD, dec_Whh_bf, HPAD, gates, G4H,
                           xp_dec + (size_t)t * G4H, T_ * G4H,
                           (const float*)nullptr, B_, G4H, HPAD);
        hipLaunchKernelGGL(lstm_cell_kernel, cgrid((size_t)B_ * HPAD), blk, 0, stream,
                           gates, h_st, c_st, h_bf, dec_h, dec_h_bf, t, T_);
    }

    // ---- attention projections ----
    hipLaunchKernelGGL(gemm_bf16_kernel, dim3(NH_PAD / 256, (B_ * S_) / 32), blk, 0, stream,
                       enc_h_bf, HPAD, Ua_w_bf, HPAD, Ua, H_,
                       (const float*)nullptr, 0, Ua_b, B_ * S_, H_, HPAD);
    hipLaunchKernelGGL(gemm_bf16_kernel, dim3(NH_PAD / 256, (B_ * T_) / 32), blk, 0, stream,
                       dec_h_bf, HPAD, Wa_w_bf, HPAD, Wa, H_,
                       (const float*)nullptr, 0, Wa_b, B_ * T_, H_, HPAD);

    // ---- scores, softmax, context ----
    hipLaunchKernelGGL(attn_scores_kernel, cgrid((size_t)B_ * T_ * S_ * 32), blk, 0, stream,
                       Ua, Wa, Va_w, Va_b, scores);
    hipLaunchKernelGGL(softmax_s_kernel, cgrid((size_t)B_ * T_), blk, 0, stream,
                       scores, attn);
    hipLaunchKernelGGL(context_kernel, cgrid((size_t)B_ * T_ * HPAD), blk, 0, stream,
                       attn, enc_h, ctx_bf);

    // ---- u = dec_h@U^T + U_b + e_en@V^T + V_b + ctx@C^T + C_b ----
    hipLaunchKernelGGL(gemm_bf16_kernel, dim3(N2M_PAD / 256, (B_ * T_) / 32), blk, 0, stream,
                       dec_h_bf, HPAD, U_w_bf, HPAD, u_buf, 2 * M_,
                       (const float*)nullptr, 0, U_b, B_ * T_, 2 * M_, HPAD);
    hipLaunchKernelGGL(gemm_bf16_kernel, dim3(N2M_PAD / 256, (B_ * T_) / 32), blk, 0, stream,
                       e_en_bf, DPAD, V_w_bf, DPAD, u_buf, 2 * M_,
                       u_buf, 2 * M_, V_b, B_ * T_, 2 * M_, DPAD);
    hipLaunchKernelGGL(gemm_bf16_kernel, dim3(N2M_PAD / 256, (B_ * T_) / 32), blk, 0, stream,
                       ctx_bf, HPAD, C_w_bf, HPAD, u_buf, 2 * M_,
                       u_buf, 2 * M_, C_b, B_ * T_, 2 * M_, HPAD);

    // ---- maxout ----
    hipLaunchKernelGGL(maxout_kernel, cgrid((size_t)B_ * T_ * HPAD), blk, 0, stream,
                       u_buf, tmax_bf);

    // ---- logits = t_max @ W_w^T + W_b  (dominant GEMM) ----
    hipLaunchKernelGGL(gemm_bf16_kernel, dim3(NV_PAD / 256, (B_ * T_) / 32), blk, 0, stream,
                       tmax_bf, HPAD, W_w_bf, HPAD, logits, VEN_,
                       (const float*)nullptr, 0, W_b, B_ * T_, VEN_, HPAD);

    // ---- log-softmax NLL + deterministic reduction ----
    hipLaunchKernelGGL(loss_row_kernel, dim3(B_ * T_), blk, 0, stream,
                       logits, y, W_b, nll);
    hipLaunchKernelGGL(loss_final_kernel, dim3(1), blk, 0, stream, nll, out);
}
